// AttentionHead_42125039239484
// MI455X (gfx1250) — matrix operand order repaired and checked by reading, not compiled, not run on hardware
//
#include <hip/hip_runtime.h>

typedef __attribute__((ext_vector_type(16))) __bf16 v16bf;
typedef __attribute__((ext_vector_type(8)))  __bf16 v8bf;
typedef __attribute__((ext_vector_type(8)))  float  v8f;
typedef __attribute__((ext_vector_type(8)))  unsigned short u16x8;
typedef __attribute__((ext_vector_type(4)))  unsigned short u16x4;
typedef __attribute__((ext_vector_type(4)))  float  f32x4;

#define HID  768
#define NSEQ 4096
#define NB   4
#define DH   64
#define SP   72   // padded LDS row stride (elements): bank = 36*lane mod 64, conflict-free

// native bf16 convert, bit pattern as u16
__device__ __forceinline__ unsigned short f2bf(float f) {
  return __builtin_bit_cast(unsigned short, (__bf16)f);
}
// fragment = two contiguous 8-element (16B) runs -> two b128 loads
__device__ __forceinline__ v16bf ld_frag(const unsigned short* p0, const unsigned short* p1) {
  v8bf lo = __builtin_bit_cast(v8bf, *(const u16x8*)p0);
  v8bf hi = __builtin_bit_cast(v8bf, *(const u16x8*)p1);
  return __builtin_shufflevector(lo, hi, 0,1,2,3,4,5,6,7,8,9,10,11,12,13,14,15);
}

// ---------------------------------------------------------------------------
// Kernel 0 (one-time, tiny): W[768][64] fp32 -> Wt[3][64][768] bf16 col-major.
// ---------------------------------------------------------------------------
__global__ __launch_bounds__(256)
void wtrans_kernel(const float* __restrict__ Wq, const float* __restrict__ Wk,
                   const float* __restrict__ Wv, unsigned short* __restrict__ Wt) {
  int idx = blockIdx.x * 256 + threadIdx.x;
  if (idx >= 3 * HID * DH) return;
  int w = idx / (HID * DH); int rem = idx - w * HID * DH;
  int k = rem >> 6, c = rem & 63;
  const float* W = (w == 0) ? Wq : ((w == 1) ? Wk : Wv);
  Wt[(size_t)w * DH * HID + (size_t)c * HID + k] = f2bf(W[rem]);
}

// ---------------------------------------------------------------------------
// Kernel 1: fused QKV projection via bf16 WMMA, f32 accumulate.
//   Q,K computed as C^T = W^T * X^T  -> row-major stores are packed u16x8
//   V   computed as C   = X * W      -> transposed [B][64][4096] stores packed
//   Qb = (x*Wq + bq) * (log2e/8)
// ---------------------------------------------------------------------------
__global__ __launch_bounds__(256)
void qkv_proj_kernel(const float* __restrict__ x,
                     const unsigned short* __restrict__ Wt,
                     const float* __restrict__ bq, const float* __restrict__ bk,
                     const float* __restrict__ bv,
                     unsigned short* __restrict__ Qb,
                     unsigned short* __restrict__ Kb,
                     unsigned short* __restrict__ Vt) {
  __shared__ alignas(16) unsigned short Xs[128 * SP];     // x tile, [row][k]
  __shared__ alignas(16) unsigned short Ws[3 * 64 * SP];  // W tiles, [col][k]
  const int tid = threadIdx.x, wave = tid >> 5, lane = tid & 31;
  const int g = lane >> 4, mn = lane & 15;
  const int row0 = blockIdx.x * 128;

  v8f CT[2][4] = {};   // Q^T, K^T frags: M=out-col, N=x-row
  v8f CV[4]    = {};   // V frags:        M=x-row,  N=out-col

  for (int kc = 0; kc < HID / 64; ++kc) {
    __syncthreads();
    for (int i = tid; i < 128 * 16; i += 256) {           // x: float4 -> bf16x4
      int r = i >> 4, c4 = i & 15;
      f32x4 f = *(const f32x4*)&x[(size_t)(row0 + r) * HID + kc * 64 + c4 * 4];
      u16x4 h; h.x = f2bf(f.x); h.y = f2bf(f.y); h.z = f2bf(f.z); h.w = f2bf(f.w);
      *(u16x4*)&Xs[r * SP + c4 * 4] = h;
    }
    for (int i = tid; i < 3 * 64 * 8; i += 256) {         // W: straight 16B copies
      int w = i >> 9; int rem = i & 511; int r = rem >> 3, ch = rem & 7;
      *(u16x8*)&Ws[w * 64 * SP + r * SP + ch * 8] =
          *(const u16x8*)&Wt[(size_t)w * DH * HID + (size_t)r * HID + kc * 64 + ch * 8];
    }
    __syncthreads();

    v16bf Ax[2], Bx[2];                                   // X as A-frag and B-frag
#pragma unroll
    for (int kh = 0; kh < 2; ++kh) {
      const unsigned short* pa = &Xs[(wave * 16 + mn) * SP + kh * 32 + 8 * g];
      Ax[kh] = ld_frag(pa, pa + 16);
      const unsigned short* pb = &Xs[(wave * 16 + mn) * SP + kh * 32 + 8 * g];
      Bx[kh] = ld_frag(pb, pb + 16);
    }
#pragma unroll
    for (int j = 0; j < 4; ++j)
#pragma unroll
      for (int kh = 0; kh < 2; ++kh) {
#pragma unroll
        for (int w = 0; w < 2; ++w) {                     // Q^T, K^T: A = W^T
          const unsigned short* p = &Ws[w * 64 * SP + (16 * j + mn) * SP + kh * 32 + 8 * g];
          v16bf Aw = ld_frag(p, p + 16);
          CT[w][j] = __builtin_amdgcn_wmma_f32_16x16x32_bf16(
              false, Aw, false, Bx[kh], (short)0, CT[w][j], false, false);
        }
        {                                                  // V: B = W
          const unsigned short* p = &Ws[2 * 64 * SP + (16 * j + mn) * SP + kh * 32 + 8 * g];
          v16bf Bw = ld_frag(p, p + 16);
          CV[j] = __builtin_amdgcn_wmma_f32_16x16x32_bf16(
              false, Ax[kh], false, Bw, (short)0, CV[j], false, false);
        }
      }
  }

  const int b_blk = row0 >> 12;
  const int seq0  = (row0 & (NSEQ - 1)) + wave * 16;
  const float QS = 0.125f * 1.44269504088896f;            // 1/sqrt(64) * log2(e)
  const size_t rowi = (size_t)(row0 + wave * 16 + mn);    // Q/K row (per lane)
#pragma unroll
  for (int j = 0; j < 4; ++j) {
    // Q/K: lane holds 8 consecutive out-cols (16j+8g .. +7) of its row
    f32x4 bq0 = *(const f32x4*)&bq[16 * j + 8 * g], bq1 = *(const f32x4*)&bq[16 * j + 8 * g + 4];
    f32x4 bk0 = *(const f32x4*)&bk[16 * j + 8 * g], bk1 = *(const f32x4*)&bk[16 * j + 8 * g + 4];
    u16x8 qp, kp;
#pragma unroll
    for (int r = 0; r < 4; ++r) {
      qp[r]     = f2bf((CT[0][j][r]     + bq0[r]) * QS);
      qp[r + 4] = f2bf((CT[0][j][r + 4] + bq1[r]) * QS);
      kp[r]     = f2bf(CT[1][j][r]     + bk0[r]);
      kp[r + 4] = f2bf(CT[1][j][r + 4] + bk1[r]);
    }
    *(u16x8*)&Qb[rowi * DH + 16 * j + 8 * g] = qp;
    *(u16x8*)&Kb[rowi * DH + 16 * j + 8 * g] = kp;
    // V: lane holds 8 consecutive seq rows of out-col 16j+mn
    float bvv = bv[16 * j + mn];
    u16x8 vp;
#pragma unroll
    for (int r = 0; r < 8; ++r) vp[r] = f2bf(CV[j][r] + bvv);
    *(u16x8*)&Vt[(size_t)b_blk * DH * NSEQ + (size_t)(16 * j + mn) * NSEQ + seq0 + 8 * g] = vp;
  }
}

// ---------------------------------------------------------------------------
// Kernel 2: flash-attention, online softmax (base 2), TRANSPOSED score tiles:
//   S^T = K * Q^T  -> each lane owns ONE query row; scores are in-lane
//   O^T = V^T * P^T -> P^T written as packed b128, read back as B-frags
// Block = 128 threads = 4 wave32; 64 query rows/block; 64-key tiles.
// ---------------------------------------------------------------------------
__global__ __launch_bounds__(128)
void flash_attn_kernel(const unsigned short* __restrict__ Qb,
                       const unsigned short* __restrict__ Kb,
                       const unsigned short* __restrict__ Vt,
                       float* __restrict__ out) {
  __shared__ alignas(16) unsigned short Ks[64 * SP];      // [key][d]
  __shared__ alignas(16) unsigned short Vs[64 * SP];      // [dim][key]
  __shared__ alignas(16) unsigned short Ps[4][16 * SP];   // per-wave P^T, [qrow][key]
  const int tid = threadIdx.x, wave = tid >> 5, lane = tid & 31;
  const int g = lane >> 4, mn = lane & 15;
  const int b  = blockIdx.x >> 6;                         // 64 blocks per batch
  const int q0 = (blockIdx.x & 63) * 64;
  const int qrow = q0 + wave * 16 + mn;                   // this lane's query row

  // loop-invariant Q^T B-fragments straight from global
  v16bf Bq[2];
  {
    const unsigned short* qp = Qb + ((size_t)b * NSEQ + qrow) * DH;
#pragma unroll
    for (int kh = 0; kh < 2; ++kh) {
      const unsigned short* p = qp + kh * 32 + 8 * g;
      Bq[kh] = ld_frag(p, p + 16);
    }
  }

  v8f O[4] = {};                                          // O^T frags: M=dim, N=qrow
  float rmax = -3.0e38f, rsum = 0.0f;                     // scalars: one row per lane

  const unsigned short* Kbase = Kb + (size_t)b * NSEQ * DH;
  const unsigned short* Vbase = Vt + (size_t)b * DH * NSEQ;

  for (int kv = 0; kv < NSEQ; kv += 64) {
    __syncthreads();
    for (int i = tid; i < 64 * 8; i += 128) {
      int r = i >> 3, ch = i & 7;
      *(u16x8*)&Ks[r * SP + ch * 8] = *(const u16x8*)&Kbase[(size_t)(kv + r) * DH + ch * 8];
      *(u16x8*)&Vs[r * SP + ch * 8] = *(const u16x8*)&Vbase[(size_t)r * NSEQ + kv + ch * 8];
    }
    __syncthreads();

    // S^T = K * Q^T : c[t] covers keys 16t..16t+15 (M) x 16 qrows (N)
    v8f c[4] = {};
#pragma unroll
    for (int t = 0; t < 4; ++t)
#pragma unroll
      for (int kh = 0; kh < 2; ++kh) {
        const unsigned short* p = &Ks[(16 * t + mn) * SP + kh * 32 + 8 * g];
        v16bf Ak = ld_frag(p, p + 16);
        c[t] = __builtin_amdgcn_wmma_f32_16x16x32_bf16(
            false, Ak, false, Bq[kh], (short)0, c[t], false, false);
      }

    // online softmax: lane's 32 scores (keys 16t + r + 8g) are all in-lane;
    // only its g-partner lane shares the row -> single shfl_xor(16)
    v8f m8 = c[0];
#pragma unroll
    for (int t = 1; t < 4; ++t)
#pragma unroll
      for (int r = 0; r < 8; ++r) m8[r] = fmaxf(m8[r], c[t][r]);
    float tmax = m8[0];
#pragma unroll
    for (int r = 1; r < 8; ++r) tmax = fmaxf(tmax, m8[r]);
    tmax = fmaxf(tmax, __shfl_xor(tmax, 16, 32));

    float mnew  = fmaxf(rmax, tmax);
    float alpha = exp2f(rmax - mnew);
    rmax = mnew;
    rsum *= alpha;
#pragma unroll
    for (int j = 0; j < 4; ++j)
#pragma unroll
      for (int r = 0; r < 8; ++r) O[j][r] *= alpha;
    float s = 0.0f;
#pragma unroll
    for (int t = 0; t < 4; ++t)
#pragma unroll
      for (int r = 0; r < 8; ++r) {
        float p = exp2f(c[t][r] - mnew); c[t][r] = p; s += p;
      }
    rsum += s;                                            // per-lane partial (g halves)

    // P^T store: lane's 8 values per t are CONSECUTIVE keys -> packed b128
#pragma unroll
    for (int t = 0; t < 4; ++t) {
      u16x8 pk;
#pragma unroll
      for (int r = 0; r < 8; ++r) pk[r] = f2bf(c[t][r]);
      *(u16x8*)&Ps[wave][mn * SP + 16 * t + 8 * g] = pk;
    }
    __syncthreads();

    // O^T += V^T * P^T : A from Vs[dim][key], B from Ps[qrow][key]
    v16bf Bp[2];
#pragma unroll
    for (int kh = 0; kh < 2; ++kh) {
      const unsigned short* p = &Ps[wave][mn * SP + kh * 32 + 8 * g];
      Bp[kh] = ld_frag(p, p + 16);
    }
#pragma unroll
    for (int j = 0; j < 4; ++j)
#pragma unroll
      for (int kh = 0; kh < 2; ++kh) {
        const unsigned short* p = &Vs[(16 * j + mn) * SP + kh * 32 + 8 * g];
        v16bf Av = ld_frag(p, p + 16);
        O[j] = __builtin_amdgcn_wmma_f32_16x16x32_bf16(
            false, Av, false, Bp[kh], (short)0, O[j], false, false);
      }
  }

  // combine the two key-half partials of this row, normalize, packed stores
  rsum += __shfl_xor(rsum, 16, 32);
  float inv = 1.0f / rsum;
  float* ob = out + ((size_t)b * NSEQ + qrow) * DH;       // lane's own row
#pragma unroll
  for (int j = 0; j < 4; ++j) {                           // dims 16j+8g .. +7
    f32x4 lo, hi;
#pragma unroll
    for (int r = 0; r < 4; ++r) { lo[r] = O[j][r] * inv; hi[r] = O[j][r + 4] * inv; }
    *(f32x4*)&ob[16 * j + 8 * g]     = lo;
    *(f32x4*)&ob[16 * j + 8 * g + 4] = hi;
  }
}

// ---------------------------------------------------------------------------
extern "C" void kernel_launch(void* const* d_in, const int* in_sizes, int n_in,
                              void* d_out, int out_size, void* d_ws, size_t ws_size,
                              hipStream_t stream) {
  const float* x  = (const float*)d_in[0];
  const float* Wq = (const float*)d_in[1];
  const float* bq = (const float*)d_in[2];
  const float* Wk = (const float*)d_in[3];
  const float* bk = (const float*)d_in[4];
  const float* Wv = (const float*)d_in[5];
  const float* bv = (const float*)d_in[6];
  float* out = (float*)d_out;

  unsigned short* Qb = (unsigned short*)d_ws;                // 2 MB
  unsigned short* Kb = Qb + (size_t)NB * NSEQ * DH;          // 2 MB
  unsigned short* Vt = Kb + (size_t)NB * NSEQ * DH;          // 2 MB, [B][64][4096]
  unsigned short* Wt = Vt + (size_t)NB * NSEQ * DH;          // 288 KB, [3][64][768]

  wtrans_kernel<<<dim3((3 * HID * DH + 255) / 256), dim3(256), 0, stream>>>(Wq, Wk, Wv, Wt);
  qkv_proj_kernel<<<dim3(NB * NSEQ / 128), dim3(256), 0, stream>>>(
      x, Wt, bq, bk, bv, Qb, Kb, Vt);
  flash_attn_kernel<<<dim3(NB * NSEQ / 64), dim3(128), 0, stream>>>(Qb, Kb, Vt, out);

  (void)in_sizes; (void)n_in; (void)out_size; (void)ws_size;
}
